// up_block_4449586118756
// MI455X (gfx1250) — matrix operand order repaired and checked by reading, not compiled, not run on hardware
//
#include <hip/hip_runtime.h>
#include <hip/hip_bf16.h>

// ---------- types ----------
typedef __attribute__((ext_vector_type(16))) __bf16   v16bf;
typedef __attribute__((ext_vector_type(8)))  float    v8f;
typedef __attribute__((ext_vector_type(4)))  unsigned u32x4;

union ABu { u32x4 u[2]; v16bf v; };

__device__ __forceinline__ unsigned short f2bf(float f) {
    unsigned u = __float_as_uint(f);
    u += 0x7fffu + ((u >> 16) & 1u);      // round-to-nearest-even
    return (unsigned short)(u >> 16);
}
__device__ __forceinline__ float bf2f(unsigned short h) {
    return __uint_as_float(((unsigned)h) << 16);
}

// ---------- fp32 -> bf16 convert ----------
__global__ void cvt_f32_bf16(const float* __restrict__ s, unsigned short* __restrict__ d, int n) {
    int i = blockIdx.x * blockDim.x + threadIdx.x;
    if (i < n) d[i] = f2bf(s[i]);
}

// ---------- GEMM1: y = x1 @ W_up.T + b_up  (M x 448, K=128), bf16 in, bf16 out ----------
// one wave = 32 rows x 64 cols (2 row-tiles x 4 WMMA col-tiles); B reused across row-tiles
__global__ __launch_bounds__(32) void gemm_up(const unsigned short* __restrict__ A,
                                              const unsigned short* __restrict__ W,
                                              const float* __restrict__ bias,
                                              unsigned short* __restrict__ Y, int M) {
    const int lane = threadIdx.x, ml = lane & 15, hi = lane >> 4;
    const int rowbase = blockIdx.x * 32;
    const int colbase = blockIdx.y * 64;
    int rm0 = rowbase + ml;      if (rm0 > M - 1) rm0 = M - 1;
    int rm1 = rowbase + 16 + ml; if (rm1 > M - 1) rm1 = M - 1;
    const unsigned short* arow0 = A + (size_t)rm0 * 128;
    const unsigned short* arow1 = A + (size_t)rm1 * 128;

    v8f zero = {0.f,0.f,0.f,0.f,0.f,0.f,0.f,0.f};
    v8f acc[2][4] = {{zero,zero,zero,zero},{zero,zero,zero,zero}};

#pragma unroll
    for (int kb = 0; kb < 128; kb += 32) {
        ABu A0, A1, B[4];
        const unsigned short* ap0 = arow0 + kb + hi * 8;
        const unsigned short* ap1 = arow1 + kb + hi * 8;
        A0.u[0] = *(const u32x4*)ap0;  A0.u[1] = *(const u32x4*)(ap0 + 16);
        A1.u[0] = *(const u32x4*)ap1;  A1.u[1] = *(const u32x4*)(ap1 + 16);
#pragma unroll
        for (int ct = 0; ct < 4; ++ct) {
            const unsigned short* bp = W + (size_t)(colbase + ct * 16 + ml) * 128 + kb + hi * 8;
            B[ct].u[0] = *(const u32x4*)bp;
            B[ct].u[1] = *(const u32x4*)(bp + 16);
        }
#pragma unroll
        for (int ct = 0; ct < 4; ++ct) {
            acc[0][ct] = __builtin_amdgcn_wmma_f32_16x16x32_bf16(false, A0.v, false, B[ct].v,
                                                                 (short)0, acc[0][ct], false, false);
            acc[1][ct] = __builtin_amdgcn_wmma_f32_16x16x32_bf16(false, A1.v, false, B[ct].v,
                                                                 (short)0, acc[1][ct], false, false);
        }
    }
#pragma unroll
    for (int rt = 0; rt < 2; ++rt) {
#pragma unroll
        for (int ct = 0; ct < 4; ++ct) {
            int n = colbase + ct * 16 + ml;
            float b = bias[n];
#pragma unroll
            for (int v = 0; v < 8; ++v) {
                int row = rowbase + rt * 16 + v + hi * 8;
                if (row < M) Y[(size_t)row * 448 + n] = f2bf(acc[rt][ct][v] + b);
            }
        }
    }
}

// ---------- build x = [up | x2] as bf16 [NH, 128] ----------
// down block follows reference's reshape(-1, 64, 2).mean(2): pairs ADJACENT CHANNELS,
// row didx[2k] supplies c<32, row didx[2k+1] supplies c>=32.
__global__ void build_x(const unsigned short* __restrict__ Y, const float* __restrict__ x2,
                        const int* __restrict__ tidx, const int* __restrict__ didx,
                        unsigned short* __restrict__ X, int NL) {
    int r = blockIdx.x, c = threadIdx.x;   // blockDim = 128
    float val;
    if (c < 64) {
        if (r < NL) {
            val = bf2f(Y[(size_t)tidx[r] * 64 + c]);
        } else {
            int k = r - NL;
            int which = (c >= 32) ? 1 : 0;
            int ysrc = didx[2 * k + which];
            int col0 = (c - which * 32) * 2;
            val = 0.5f * (bf2f(Y[(size_t)ysrc * 64 + col0]) +
                          bf2f(Y[(size_t)ysrc * 64 + col0 + 1]));
        }
    } else {
        val = x2[(size_t)r * 64 + (c - 64)];
    }
    X[(size_t)r * 128 + c] = f2bf(val);
}

// ---------- onering conv GEMM (templated on C for full unroll) ----------
// out[r,n] = sum_j sum_k S[neigh[7r+j],k]*W[n, j*C+k] + bias[n]
// one wave = 32 rows (2 row-tiles); fused gather + per-channel sum/sumsq for BN (tail-masked)
template <int C>
__global__ __launch_bounds__(32) void conv_gemm(const unsigned short* __restrict__ S,
                                                const int* __restrict__ neigh,
                                                const unsigned short* __restrict__ W,
                                                const float* __restrict__ bias,
                                                float* __restrict__ out,
                                                float* __restrict__ sums,   // [0:64)=sum, [64:128)=sumsq
                                                int M) {
    const int lane = threadIdx.x, ml = lane & 15, hi = lane >> 4;
    const int rowbase = blockIdx.x * 32;
    int rm0 = rowbase + ml;      if (rm0 > M - 1) rm0 = M - 1;
    int rm1 = rowbase + 16 + ml; if (rm1 > M - 1) rm1 = M - 1;
    constexpr int K7 = 7 * C;

    int idx0[7], idx1[7];
#pragma unroll
    for (int j = 0; j < 7; ++j) {
        idx0[j] = neigh[(size_t)rm0 * 7 + j];
        idx1[j] = neigh[(size_t)rm1 * 7 + j];
    }

    v8f zero = {0.f,0.f,0.f,0.f,0.f,0.f,0.f,0.f};
    v8f acc[2][4] = {{zero,zero,zero,zero},{zero,zero,zero,zero}};

#pragma unroll
    for (int j = 0; j < 7; ++j) {
        const unsigned short* arow0 = S + (size_t)idx0[j] * C;
        const unsigned short* arow1 = S + (size_t)idx1[j] * C;
#pragma unroll
        for (int kb = 0; kb < C; kb += 32) {
            ABu A0, A1, B[4];
            const unsigned short* ap0 = arow0 + kb + hi * 8;
            const unsigned short* ap1 = arow1 + kb + hi * 8;
            A0.u[0] = *(const u32x4*)ap0;  A0.u[1] = *(const u32x4*)(ap0 + 16);
            A1.u[0] = *(const u32x4*)ap1;  A1.u[1] = *(const u32x4*)(ap1 + 16);
#pragma unroll
            for (int ct = 0; ct < 4; ++ct) {
                const unsigned short* bp = W + (size_t)(ct * 16 + ml) * K7 + j * C + kb + hi * 8;
                B[ct].u[0] = *(const u32x4*)bp;
                B[ct].u[1] = *(const u32x4*)(bp + 16);
            }
#pragma unroll
            for (int ct = 0; ct < 4; ++ct) {
                acc[0][ct] = __builtin_amdgcn_wmma_f32_16x16x32_bf16(false, A0.v, false, B[ct].v,
                                                                     (short)0, acc[0][ct], false, false);
                acc[1][ct] = __builtin_amdgcn_wmma_f32_16x16x32_bf16(false, A1.v, false, B[ct].v,
                                                                     (short)0, acc[1][ct], false, false);
            }
        }
    }
#pragma unroll
    for (int ct = 0; ct < 4; ++ct) {
        int n = ct * 16 + ml;
        float b = bias[n];
        float s = 0.f, q = 0.f;
#pragma unroll
        for (int rt = 0; rt < 2; ++rt) {
#pragma unroll
            for (int v = 0; v < 8; ++v) {
                int row = rowbase + rt * 16 + v + hi * 8;
                if (row < M) {
                    float val = acc[rt][ct][v] + b;
                    out[(size_t)row * 64 + n] = val;
                    s += val; q += val * val;
                }
            }
        }
        atomicAdd(&sums[n], s);
        atomicAdd(&sums[64 + n], q);
    }
}

// ---------- BN finalize: mu, rstd from sum/sumsq (biased variance) ----------
__global__ void bn_finalize(const float* __restrict__ sums, float* __restrict__ musig, float n) {
    int c = threadIdx.x;   // 64 threads
    float mu = sums[c] / n;
    float var = sums[64 + c] / n - mu * mu;
    musig[c] = mu;
    musig[64 + c] = rsqrtf(var + 1e-5f);
}

// ---------- BN apply + LeakyReLU(0.2); bf16 or fp32 output ----------
template <bool BF16OUT>
__global__ void bn_apply(const float* __restrict__ h, const float* __restrict__ musig,
                         const float* __restrict__ g, const float* __restrict__ be,
                         void* __restrict__ out, int total) {
    int i = blockIdx.x * blockDim.x + threadIdx.x;
    if (i >= total) return;
    int c = i & 63;
    float y = g[c] * (h[i] - musig[c]) * musig[64 + c] + be[c];
    y = (y >= 0.f) ? y : 0.2f * y;
    if (BF16OUT) ((unsigned short*)out)[i] = f2bf(y);
    else         ((float*)out)[i] = y;
}

extern "C" void kernel_launch(void* const* d_in, const int* in_sizes, int n_in,
                              void* d_out, int out_size, void* d_ws, size_t ws_size,
                              hipStream_t stream) {
    const float* x1   = (const float*)d_in[0];
    const float* x2   = (const float*)d_in[1];
    const float* W_up = (const float*)d_in[2];
    const float* b_up = (const float*)d_in[3];
    const float* W1   = (const float*)d_in[4];
    const float* b1   = (const float*)d_in[5];
    const float* g1   = (const float*)d_in[6];
    const float* be1  = (const float*)d_in[7];
    const float* W2   = (const float*)d_in[8];
    const float* b2   = (const float*)d_in[9];
    const float* g2   = (const float*)d_in[10];
    const float* be2  = (const float*)d_in[11];
    const int* neigh  = (const int*)d_in[12];
    const int* tidx   = (const int*)d_in[13];
    const int* didx   = (const int*)d_in[14];
    float* outp = (float*)d_out;

    const int NL = in_sizes[0] / 128;   // 40962
    const int NH = in_sizes[1] / 64;    // 163842

    // workspace carve-out (256B aligned)
    char* p = (char*)d_ws;
    auto carve = [&](size_t bytes) -> char* {
        char* r = p; p += (bytes + 255) & ~(size_t)255; return r;
    };
    unsigned short* x1b  = (unsigned short*)carve((size_t)NL * 128 * 2);
    unsigned short* wupb = (unsigned short*)carve((size_t)448 * 128 * 2);
    unsigned short* w1b  = (unsigned short*)carve((size_t)64 * 896 * 2);
    unsigned short* w2b  = (unsigned short*)carve((size_t)64 * 448 * 2);
    unsigned short* yb   = (unsigned short*)carve((size_t)7 * NL * 64 * 2);
    unsigned short* xb   = (unsigned short*)carve((size_t)NH * 128 * 2);
    unsigned short* h1b  = (unsigned short*)carve((size_t)NH * 64 * 2);
    float*          hraw = (float*)carve((size_t)NH * 64 * 4);
    float*          stats = (float*)carve(256 * 4);   // sum[64], sumsq[64], mu[64], rstd[64]
    float*          musig = stats + 128;

    // 1) bf16 conversions
    {
        int n;
        n = NL * 128;  cvt_f32_bf16<<<(n + 255) / 256, 256, 0, stream>>>(x1,   x1b,  n);
        n = 448 * 128; cvt_f32_bf16<<<(n + 255) / 256, 256, 0, stream>>>(W_up, wupb, n);
        n = 64 * 896;  cvt_f32_bf16<<<(n + 255) / 256, 256, 0, stream>>>(W1,   w1b,  n);
        n = 64 * 448;  cvt_f32_bf16<<<(n + 255) / 256, 256, 0, stream>>>(W2,   w2b,  n);
    }

    // 2) y = x1 @ W_up.T + b_up  -> bf16 [7*NL, 64] view
    {
        dim3 grid((NL + 31) / 32, 7);
        gemm_up<<<grid, 32, 0, stream>>>(x1b, wupb, b_up, yb, NL);
    }

    // 3) x = [up | x2] bf16 [NH,128]
    build_x<<<NH, 128, 0, stream>>>(yb, x2, tidx, didx, xb, NL);

    // 4) conv1 + BN stats
    hipMemsetAsync(stats, 0, 128 * sizeof(float), stream);
    conv_gemm<128><<<(NH + 31) / 32, 32, 0, stream>>>(xb, neigh, w1b, b1, hraw, stats, NH);
    bn_finalize<<<1, 64, 0, stream>>>(stats, musig, (float)NH);
    bn_apply<true><<<(NH * 64 + 255) / 256, 256, 0, stream>>>(hraw, musig, g1, be1, h1b, NH * 64);

    // 5) conv2 + BN stats -> d_out
    hipMemsetAsync(stats, 0, 128 * sizeof(float), stream);
    conv_gemm<64><<<(NH + 31) / 32, 32, 0, stream>>>(h1b, neigh, w2b, b2, hraw, stats, NH);
    bn_finalize<<<1, 64, 0, stream>>>(stats, musig, (float)NH);
    bn_apply<false><<<(NH * 64 + 255) / 256, 256, 0, stream>>>(hraw, musig, g2, be2, outp, NH * 64);
}
